// MultiHeadAttention_32100585570877
// MI455X (gfx1250) — matrix operand order repaired and checked by reading, not compiled, not run on hardware
//
#include <hip/hip_runtime.h>

// ---------------------------------------------------------------------------
// MultiHeadAttention forward for MI455X (gfx1250, wave32, WMMA).
// B=4, S=2048, D_MODEL=2048, H=16, D_K=128.
// Phase 1: QKV projection GEMMs (f32 in -> bf16 WMMA -> bf16 out in ws).
//          Q,K stored [B,H,S,Dk] (d contiguous); V stored transposed [B,H,Dk,S].
// Phase 2: flash attention per (64-row q tile, b*h), online softmax,
//          double-buffered K/V tiles streamed with async LDS loads.
// ---------------------------------------------------------------------------

typedef __attribute__((ext_vector_type(16))) __bf16 v16bf;
typedef __attribute__((ext_vector_type(8)))  __bf16 bf16x8;
typedef __attribute__((ext_vector_type(8)))  float  v8f;
typedef __attribute__((ext_vector_type(4)))  int    v4i;

#define S_LEN   2048
#define D_MODEL 2048
#define NHEAD   16
#define DK      128
#define BATCH   4

// ---- async global->LDS (CDNA5) with graceful fallback ---------------------
#if defined(__has_builtin)
#  if __has_builtin(__builtin_amdgcn_global_load_async_to_lds_b128)
#    define HAVE_ASYNC_LOAD 1
#  endif
#  if __has_builtin(__builtin_amdgcn_s_wait_asynccnt)
#    define HAVE_WAIT_ASYNC 1
#  endif
#endif

// builtin parameter types (from compiler diagnostic): v4i in AS1 / AS3
typedef __attribute__((address_space(1))) v4i g1v4i;
typedef __attribute__((address_space(3))) v4i l3v4i;

__device__ __forceinline__ void copy16_to_lds(const void* g, void* l) {
#if HAVE_ASYNC_LOAD
    // GLOBAL_LOAD_ASYNC_TO_LDS_B128: per-lane global addr -> per-lane LDS addr
    __builtin_amdgcn_global_load_async_to_lds_b128((g1v4i*)g, (l3v4i*)l, 0, 0);
#else
    *(bf16x8*)l = *(const bf16x8*)g;
#endif
}

__device__ __forceinline__ void wait_async_all() {
#if HAVE_ASYNC_LOAD
#  if HAVE_WAIT_ASYNC
    __builtin_amdgcn_s_wait_asynccnt(0);
#  else
    asm volatile("s_wait_asynccnt 0x0" ::: "memory");
#  endif
#endif
}

// ---- WMMA helper ----------------------------------------------------------
__device__ __forceinline__ v8f wmma_bf16(v16bf a, v16bf b, v8f c) {
    // D = A(16x32 bf16) * B(32x16 bf16) + C(16x16 f32)
    return __builtin_amdgcn_wmma_f32_16x16x32_bf16(
        /*neg_a=*/false, a, /*neg_b=*/false, b,
        /*c_mod=*/(short)0, c, /*reuse_a=*/false, /*reuse_b=*/false);
}

// A fragment (16x32, 16-bit): lane m = lane&15; lanes 0-15 hold K 0..7/16..23,
// lanes 16-31 hold K 8..15/24..31  -> two 16B LDS loads per lane.
__device__ __forceinline__ v16bf ld_afrag(const __bf16* tile, int row0, int ld, int koff) {
    const int lane = threadIdx.x & 31;
    const int m    = lane & 15;
    const int sel  = lane >> 4;               // 0 or 1
    const __bf16* p = tile + (row0 + m) * ld + koff + sel * 8;
    bf16x8 lo = *(const bf16x8*)p;            // K = koff + sel*8 + [0..7]
    bf16x8 hi = *(const bf16x8*)(p + 16);     // K = koff + 16 + sel*8 + [0..7]
    v16bf r;
#pragma unroll
    for (int i = 0; i < 8; ++i) { r[i] = lo[i]; r[i + 8] = hi[i]; }
    return r;
}

// B fragment (32x16, 16-bit): lane n = lane&15; lanes 0-15 hold K 0..7/16..23,
// lanes 16-31 hold K 8..15/24..31. tile layout is [n][k] (k contiguous, stride ld).
__device__ __forceinline__ v16bf ld_bfrag(const __bf16* tile, int col0, int ld, int koff) {
    const int lane = threadIdx.x & 31;
    const int n    = lane & 15;
    const int sel  = lane >> 4;
    const __bf16* p = tile + (col0 + n) * ld + koff + sel * 8;
    bf16x8 lo = *(const bf16x8*)p;
    bf16x8 hi = *(const bf16x8*)(p + 16);
    v16bf r;
#pragma unroll
    for (int i = 0; i < 8; ++i) { r[i] = lo[i]; r[i + 8] = hi[i]; }
    return r;
}

// ---------------------------------------------------------------------------
// Phase 1: C[8192,2048] = X[8192,2048] @ W[2048,2048] + b, bf16 output.
// blockIdx.z: 0 = Q, 1 = K, 2 = V(transposed store).
// Workgroup tile 128x128, K-stage 64; 8 waves as 4(M) x 2(N); 16 WMMA/stage.
// ---------------------------------------------------------------------------
__global__ __launch_bounds__(256)
void qkv_proj_kernel(const float* __restrict__ xq, const float* __restrict__ xk,
                     const float* __restrict__ xv,
                     const float* __restrict__ wq, const float* __restrict__ wk,
                     const float* __restrict__ wv,
                     const float* __restrict__ bq, const float* __restrict__ bk,
                     const float* __restrict__ bv,
                     __bf16* __restrict__ qo, __bf16* __restrict__ ko,
                     __bf16* __restrict__ vo)
{
    __shared__ __attribute__((aligned(16))) __bf16 At[128 * 64];   // [m][k]
    __shared__ __attribute__((aligned(16))) __bf16 Bt[128 * 64];   // [n][k] (W^T tile)

    const int z = blockIdx.z;
    const float* x    = (z == 0) ? xq : (z == 1) ? xk : xv;
    const float* w    = (z == 0) ? wq : (z == 1) ? wk : wv;
    const float* bias = (z == 0) ? bq : (z == 1) ? bk : bv;
    __bf16* out       = (z == 0) ? qo : (z == 1) ? ko : vo;
    const int mode    = (z == 2) ? 1 : 0;

    const int t    = threadIdx.x;
    const int lane = t & 31;
    const int wid  = t >> 5;
    const int wm   = wid & 3;      // 4 wave rows  (32 M each)
    const int wn   = wid >> 2;     // 2 wave cols  (64 N each)
    const int m0   = blockIdx.y * 128;
    const int n0   = blockIdx.x * 128;

    v8f acc[2][4];
#pragma unroll
    for (int i = 0; i < 2; ++i)
#pragma unroll
        for (int j = 0; j < 4; ++j) acc[i][j] = v8f{};

    const int ar   = t >> 1;             // A stage: row 0..127
    const int ac0  = (t & 1) * 32;       //          32 floats each
    const int wcol = t & 127;            // W stage: n col 0..127
    const int wkg  = (t >> 7) * 32;      //          32 k rows each

    for (int k0 = 0; k0 < D_MODEL; k0 += 64) {
        __syncthreads();
        // stage A tile (f32 -> bf16), packed converts
        {
            const float* xp = x + (size_t)(m0 + ar) * D_MODEL + k0 + ac0;
#pragma unroll
            for (int i = 0; i < 32; i += 4) {
                float4 v = *(const float4*)(xp + i);
                At[ar * 64 + ac0 + i + 0] = (__bf16)v.x;
                At[ar * 64 + ac0 + i + 1] = (__bf16)v.y;
                At[ar * 64 + ac0 + i + 2] = (__bf16)v.z;
                At[ar * 64 + ac0 + i + 3] = (__bf16)v.w;
            }
        }
        // stage W tile transposed: Bt[n][k] = W[k][n] (coalesced across threads)
        {
#pragma unroll
            for (int i = 0; i < 32; ++i) {
                float val = w[(size_t)(k0 + wkg + i) * D_MODEL + n0 + wcol];
                Bt[wcol * 64 + wkg + i] = (__bf16)val;
            }
        }
        __syncthreads();

#pragma unroll
        for (int ks = 0; ks < 64; ks += 32) {
            v16bf a0 = ld_afrag(At, wm * 32 + 0,  64, ks);
            v16bf a1 = ld_afrag(At, wm * 32 + 16, 64, ks);
#pragma unroll
            for (int nf = 0; nf < 4; ++nf) {
                v16bf b = ld_bfrag(Bt, wn * 64 + nf * 16, 64, ks);
                acc[0][nf] = wmma_bf16(a0, b, acc[0][nf]);
                acc[1][nf] = wmma_bf16(a1, b, acc[1][nf]);
            }
        }
    }

    // epilogue: + bias, bf16 store (Q/K: [B,H,S,Dk]; V: [B,H,Dk,S])
    const int nIdx = lane & 15;
    const int sel  = lane >> 4;
#pragma unroll
    for (int mf = 0; mf < 2; ++mf) {
#pragma unroll
        for (int nf = 0; nf < 4; ++nf) {
#pragma unroll
            for (int r = 0; r < 8; ++r) {
                const int gm = m0 + wm * 32 + mf * 16 + r + sel * 8;
                const int gn = n0 + wn * 64 + nf * 16 + nIdx;
                const float val = acc[mf][nf][r] + bias[gn];
                const int bb = gm >> 11, ss = gm & 2047;
                const int hh = gn >> 7,  dd = gn & 127;
                const int idx = (mode == 0)
                    ? (((bb * NHEAD + hh) * S_LEN + ss) * DK + dd)
                    : (((bb * NHEAD + hh) * DK + dd) * S_LEN + ss);
                out[idx] = (__bf16)val;
            }
        }
    }
}

// ---------------------------------------------------------------------------
// Phase 2: flash attention. grid = (S/64, B*H), block = 256 (8 waves).
// Q tile 64x128 resident in LDS; double-buffered 64-key K/V blocks streamed
// with async global->LDS loads (ASYNCcnt); online softmax.
// Wave layout: wm = wave&3 -> 16 query rows; wn = wave>>2 ->
//   S phase: 32 key cols, O phase: 64 d cols (4 frags).
// ---------------------------------------------------------------------------
__global__ __launch_bounds__(256)
void flash_attn_kernel(const __bf16* __restrict__ Qb, const __bf16* __restrict__ Kb,
                       const __bf16* __restrict__ Vt, float* __restrict__ out)
{
    __shared__ __attribute__((aligned(16))) __bf16 Qs[64 * 128];      // [q][d]
    __shared__ __attribute__((aligned(16))) __bf16 Ks[2][64 * 128];   // [k][d]
    __shared__ __attribute__((aligned(16))) __bf16 Vs[2][128 * 64];   // [d][k]
    __shared__ __attribute__((aligned(16))) float  Ss[64 * 64];       // scores
    __shared__ __attribute__((aligned(16))) __bf16 Ps[64 * 64];       // probs
    __shared__ float rmax[64], rsum[64], ralpha[64];

    const int t    = threadIdx.x;
    const int lane = t & 31;
    const int wid  = t >> 5;
    const int wm   = wid & 3;
    const int wn   = wid >> 2;
    const int q0   = blockIdx.x * 64;
    const int bh   = blockIdx.y;
    const int bb   = bh >> 4, hh = bh & 15;
    const float scale = 0.08838834764831845f;   // 1/sqrt(128)

    // per-thread staging coordinates
    const int krow = t >> 2, kc0 = (t & 3) * 32;   // K/Q: 4 threads x 32 bf16 per row
    const int vd   = t >> 1, vc0 = (t & 1) * 32;   // V: 2 threads x 32 bf16 per d-row

    const __bf16* Kbase = Kb + (size_t)bh * S_LEN * DK;
    const __bf16* Vbase = Vt + (size_t)bh * DK * S_LEN;

    // issue async stage of Q tile and first K/V block
    {
        const __bf16* qsrc = Qb + ((size_t)bh * S_LEN + q0 + krow) * DK + kc0;
#pragma unroll
        for (int i = 0; i < 32; i += 8)
            copy16_to_lds(qsrc + i, Qs + krow * 128 + kc0 + i);
        const __bf16* ksrc = Kbase + (size_t)(0 + krow) * DK + kc0;
        const __bf16* vsrc = Vbase + (size_t)vd * S_LEN + 0 + vc0;
#pragma unroll
        for (int i = 0; i < 32; i += 8) {
            copy16_to_lds(ksrc + i, Ks[0] + krow * 128 + kc0 + i);
            copy16_to_lds(vsrc + i, Vs[0] + vd * 64 + vc0 + i);
        }
    }
    if (t < 64) { rmax[t] = -1e30f; rsum[t] = 0.0f; }

    v8f acco[4];
#pragma unroll
    for (int i = 0; i < 4; ++i) acco[i] = v8f{};

    const int nIdx = lane & 15;
    const int sel  = lane >> 4;

    wait_async_all();
    __syncthreads();

    int buf = 0;
    for (int kb = 0; kb < S_LEN; kb += 64) {
        // prefetch next K/V block into the other buffer (overlaps compute)
        if (kb + 64 < S_LEN) {
            const int nb = buf ^ 1;
            const __bf16* ksrc = Kbase + (size_t)(kb + 64 + krow) * DK + kc0;
            const __bf16* vsrc = Vbase + (size_t)vd * S_LEN + (kb + 64) + vc0;
#pragma unroll
            for (int i = 0; i < 32; i += 8) {
                copy16_to_lds(ksrc + i, Ks[nb] + krow * 128 + kc0 + i);
                copy16_to_lds(vsrc + i, Vs[nb] + vd * 64 + vc0 + i);
            }
        }

        // S = (Q @ K^T) * scale  -> Ss
#pragma unroll
        for (int nf = 0; nf < 2; ++nf) {
            v8f s = v8f{};
#pragma unroll
            for (int ks = 0; ks < 128; ks += 32) {
                v16bf a = ld_afrag(Qs, wm * 16, 128, ks);
                v16bf b = ld_bfrag(Ks[buf], wn * 32 + nf * 16, 128, ks);
                s = wmma_bf16(a, b, s);
            }
#pragma unroll
            for (int r = 0; r < 8; ++r)
                Ss[(wm * 16 + r + sel * 8) * 64 + wn * 32 + nf * 16 + nIdx] = s[r] * scale;
        }
        __syncthreads();

        // online softmax, one thread per query row
        if (t < 64) {
            float m_old = rmax[t];
            float mx = m_old;
            const float* srow = Ss + t * 64;
#pragma unroll 8
            for (int j = 0; j < 64; ++j) mx = fmaxf(mx, srow[j]);
            const float alpha = __expf(m_old - mx);
            float sum = rsum[t] * alpha;
            __bf16* prow = Ps + t * 64;
#pragma unroll 8
            for (int j = 0; j < 64; ++j) {
                const float p = __expf(srow[j] - mx);
                sum += p;
                prow[j] = (__bf16)p;
            }
            rmax[t] = mx; rsum[t] = sum; ralpha[t] = alpha;
        }
        __syncthreads();

        // rescale O accumulators, then O += P @ V
#pragma unroll
        for (int df = 0; df < 4; ++df) {
#pragma unroll
            for (int r = 0; r < 8; ++r)
                acco[df][r] *= ralpha[wm * 16 + r + sel * 8];
        }
#pragma unroll
        for (int ks = 0; ks < 64; ks += 32) {
            v16bf a = ld_afrag(Ps, wm * 16, 64, ks);
#pragma unroll
            for (int df = 0; df < 4; ++df) {
                v16bf b = ld_bfrag(Vs[buf], wn * 64 + df * 16, 64, ks);
                acco[df] = wmma_bf16(a, b, acco[df]);
            }
        }

        // next buffer must be fully landed in LDS before anyone reads it;
        // barrier also closes out all reads of the current buffer.
        wait_async_all();
        __syncthreads();
        buf ^= 1;
    }

    // final normalize + f32 store to [B,S,H*Dk]
#pragma unroll
    for (int r = 0; r < 8; ++r) {
        const int row = wm * 16 + r + sel * 8;
        const float inv = 1.0f / rsum[row];
#pragma unroll
        for (int df = 0; df < 4; ++df) {
            const int col = hh * DK + wn * 64 + df * 16 + nIdx;
            out[((size_t)bb * S_LEN + q0 + row) * D_MODEL + col] = acco[df][r] * inv;
        }
    }
}

// ---------------------------------------------------------------------------
extern "C" void kernel_launch(void* const* d_in, const int* in_sizes, int n_in,
                              void* d_out, int out_size, void* d_ws, size_t ws_size,
                              hipStream_t stream)
{
    (void)in_sizes; (void)n_in; (void)out_size; (void)ws_size;
    const float* xq = (const float*)d_in[0];
    const float* xk = (const float*)d_in[1];
    const float* xv = (const float*)d_in[2];
    const float* Wq = (const float*)d_in[3];
    const float* bq = (const float*)d_in[4];
    const float* Wk = (const float*)d_in[5];
    const float* bk = (const float*)d_in[6];
    const float* Wv = (const float*)d_in[7];
    const float* bv = (const float*)d_in[8];

    const size_t QSZ = (size_t)BATCH * S_LEN * D_MODEL;   // elements per tensor
    __bf16* qb = (__bf16*)d_ws;
    __bf16* kb = qb + QSZ;
    __bf16* vt = kb + QSZ;                                 // total 96 MB bf16

    dim3 gproj(D_MODEL / 128, (BATCH * S_LEN) / 128, 3);
    qkv_proj_kernel<<<gproj, 256, 0, stream>>>(xq, xk, xv, Wq, Wk, Wv,
                                               bq, bk, bv, qb, kb, vt);

    dim3 gattn(S_LEN / 64, BATCH * NHEAD);
    flash_attn_kernel<<<gattn, 256, 0, stream>>>(qb, kb, vt, (float*)d_out);
}
